// IDCT_1133871366814
// MI455X (gfx1250) — matrix operand order repaired and checked by reading, not compiled, not run on hardware
//
#include <hip/hip_runtime.h>

// ============================================================================
// IDCT (DCT-III) as a dense f16 WMMA GEMM:  Y[B,N] = X[B,N] * C[N,N]
//   C[n,k] = w_n * cos(pi*n*(2k+1)/(2N)),  w_0=1, w_n>0 = 2
// Stored transposed, pre-scaled: Ct[k][n] (row-major, f16).
//
// GEMM: block tile 128x128, 8 wave32s (4Mx2N), wave tile 32x64 (2x4 WMMA accs),
// TILE_K=64 with double-buffered LDS (one barrier per 16 v_wmma) and async
// global->LDS staging (GLOBAL_LOAD_ASYNC_TO_LDS_B128, ASYNCcnt path).
//
// Workspace layout (needs >= 64 MB):
//   [0,    32MB) : Xh = f16 copy of X    [B][N]
//   [32MB, 64MB) : Ct = f16 coefficients [N][N]
// ============================================================================

typedef __attribute__((ext_vector_type(16))) _Float16 v16h;
typedef __attribute__((ext_vector_type(8)))  _Float16 v8h;
typedef __attribute__((ext_vector_type(8)))  float    v8f;
typedef __attribute__((ext_vector_type(4)))  int      v4i;

#define TILE_K     64
#define LDS_STRIDE 72            // 64 halfs + 8 pad -> 144B rows, conflict-free b128
#define TILE_ELEMS (128 * LDS_STRIDE)

#if defined(__AMDGCN__) && __has_builtin(__builtin_amdgcn_global_load_async_to_lds_b128)
#define HAVE_ASYNC_LDS 1
// Builtin signature (from hipcc diagnostic): param0 = AS1 (global) v4i*,
// param1 = AS3 (LDS) v4i*, then two integer immediates (offset, cpol).
typedef v4i __attribute__((address_space(1)))* g4i_p;
typedef v4i __attribute__((address_space(3)))* l4i_p;
#else
#define HAVE_ASYNC_LDS 0
#endif

// ---------------------------------------------------------------- X -> f16
__global__ __launch_bounds__(256) void cvt_x_f16(const float* __restrict__ x,
                                                 _Float16* __restrict__ xh,
                                                 int total) {
    int i = blockIdx.x * 256 + threadIdx.x;
    if (i < total) xh[i] = (_Float16)x[i];
}

// ------------------------------------------- Ct[k][n] = w_n * cos(pi*n*(2k+1)/(2N))
__global__ __launch_bounds__(256) void gen_ct_f16(_Float16* __restrict__ ct,
                                                  int n_dim, float ang_scale) {
    int n = blockIdx.x * 256 + threadIdx.x;   // reduction index (column of Ct)
    int k = blockIdx.y;                       // output index   (row of Ct)
    if (n >= n_dim) return;
    unsigned prod = (unsigned)n * (unsigned)(2 * k + 1);
    unsigned m    = prod & (unsigned)(4 * n_dim - 1);   // mod 4N (N is pow2)
    float w   = (n == 0) ? 1.0f : 2.0f;
    float val = w * cosf((float)m * ang_scale);          // ang_scale = pi/(2N)
    ct[(size_t)k * n_dim + n] = (_Float16)val;
}

// ---- stage one 128 x 64-half tile (g points at [row0][k0]) into LDS tile s
__device__ __forceinline__ void stage_tile(const _Float16* __restrict__ g,
                                           _Float16* s, int tid, int n_dim) {
    const int row = tid >> 3;          // 0..31 (+32*rr)
    const int col = (tid & 7) * 8;     // 0,8,...,56 halfs (16B chunks)
#pragma unroll
    for (int rr = 0; rr < 4; rr++) {
        int r = row + rr * 32;
#if HAVE_ASYNC_LDS
        __builtin_amdgcn_global_load_async_to_lds_b128(
            (g4i_p)(g + (size_t)r * n_dim + col),
            (l4i_p)(s + r * LDS_STRIDE + col),
            0, 0);
#else
        *(v8h*)(s + r * LDS_STRIDE + col) =
            *(const v8h*)(g + (size_t)r * n_dim + col);
#endif
    }
}

__device__ __forceinline__ void wait_async_all() {
#if HAVE_ASYNC_LDS
#if __has_builtin(__builtin_amdgcn_s_wait_asynccnt)
    __builtin_amdgcn_s_wait_asynccnt(0);
#else
    asm volatile("s_wait_asynccnt 0" ::: "memory");
#endif
#endif
}

// ---------------------------------------------------------------- GEMM core
__global__ __launch_bounds__(256) void idct_wmma_gemm(const _Float16* __restrict__ Xh,
                                                      const _Float16* __restrict__ Ct,
                                                      float* __restrict__ Y,
                                                      int n_dim) {
    __shared__ __align__(16) _Float16 sA[2 * TILE_ELEMS];  // 2 x 18 KB
    __shared__ __align__(16) _Float16 sB[2 * TILE_ELEMS];  // 2 x 18 KB

    const int tid  = threadIdx.x;
    const int lane = tid & 31;
    const int wid  = tid >> 5;
    const int wm   = wid & 3;            // wave row    (0..3)
    const int wn   = wid >> 2;           // wave column (0..1)
    const int m0   = blockIdx.y * 128;   // batch-row base
    const int n0   = blockIdx.x * 128;   // output-col base

    const _Float16* gA = Xh + (size_t)m0 * n_dim;   // tile row bases
    const _Float16* gB = Ct + (size_t)n0 * n_dim;

    v8f acc[2][4] = {};
    union Frag { v16h v; v8h h[2]; };

    const int fr = lane & 15;            // fragment row (A: M, B: N)
    const int ka = (lane >> 4) * 8;      // A: K interleave {0-7,16-23}/{8-15,24-31}
    const int kb = (lane >> 4) * 16;     // unused; B loads use ka, same interleave as A

    // ---- prologue: fill buffer 0 with k0 = 0
    stage_tile(gA, sA, tid, n_dim);
    stage_tile(gB, sB, tid, n_dim);
    wait_async_all();
    __syncthreads();

    int buf = 0;
    for (int k0 = 0; k0 < n_dim; k0 += TILE_K) {
        // ---- prefetch next K-slab into the other buffer (async -> LDS)
        if (k0 + TILE_K < n_dim) {
            int nb = buf ^ 1;
            stage_tile(gA + (k0 + TILE_K), sA + nb * TILE_ELEMS, tid, n_dim);
            stage_tile(gB + (k0 + TILE_K), sB + nb * TILE_ELEMS, tid, n_dim);
        }

        // ---- compute from current buffer: 2 K-chunks of 32 -> 16 v_wmma
        const _Float16* sAb = sA + buf * TILE_ELEMS;
        const _Float16* sBb = sB + buf * TILE_ELEMS;
#pragma unroll
        for (int kc = 0; kc < TILE_K; kc += 32) {
            Frag a[2], b[4];
#pragma unroll
            for (int tm = 0; tm < 2; tm++) {
                int r = wm * 32 + tm * 16 + fr;
                a[tm].h[0] = *(const v8h*)(sAb + r * LDS_STRIDE + kc + ka);
                a[tm].h[1] = *(const v8h*)(sAb + r * LDS_STRIDE + kc + 16 + ka);
            }
#pragma unroll
            for (int tn = 0; tn < 4; tn++) {
                int c = wn * 64 + tn * 16 + fr;
                b[tn].h[0] = *(const v8h*)(sBb + c * LDS_STRIDE + kc + ka);
                b[tn].h[1] = *(const v8h*)(sBb + c * LDS_STRIDE + kc + 16 + ka);
            }
#pragma unroll
            for (int tm = 0; tm < 2; tm++)
#pragma unroll
                for (int tn = 0; tn < 4; tn++)
                    acc[tm][tn] = __builtin_amdgcn_wmma_f32_16x16x32_f16(
                        false, a[tm].v, false, b[tn].v,
                        (short)0, acc[tm][tn], false, false);
        }

        // ---- our async loads landed; one barrier per K=64 slab
        wait_async_all();
        __syncthreads();
        buf ^= 1;
    }

    // ---- store: C/D layout — VGPR i: lanes 0-15 row=i, lanes 16-31 row=8+i
    const int rlo = (lane >> 4) * 8;
    const int cc  = lane & 15;
#pragma unroll
    for (int tm = 0; tm < 2; tm++) {
#pragma unroll
        for (int tn = 0; tn < 4; tn++) {
            int row = m0 + wm * 32 + tm * 16 + rlo;
            int col = n0 + wn * 64 + tn * 16 + cc;
#pragma unroll
            for (int i = 0; i < 8; i++)
                Y[(size_t)(row + i) * n_dim + col] = acc[tm][tn][i];
        }
    }
}

// ============================================================================
extern "C" void kernel_launch(void* const* d_in, const int* in_sizes, int n_in,
                              void* d_out, int out_size, void* d_ws, size_t ws_size,
                              hipStream_t stream) {
    const float* x = (const float*)d_in[0];
    // d_in[1] = expk [N,2] (unused: coefficients regenerated exactly on device)

    const int n_dim   = in_sizes[1] / 2;          // 4096
    const int b_dim   = in_sizes[0] / n_dim;      // 4096
    const int total_x = b_dim * n_dim;

    _Float16* Xh = (_Float16*)d_ws;
    _Float16* Ct = (_Float16*)((char*)d_ws + (size_t)b_dim * n_dim * sizeof(_Float16));
    float*    Y  = (float*)d_out;

    cvt_x_f16<<<(total_x + 255) / 256, 256, 0, stream>>>(x, Xh, total_x);

    dim3 gct(n_dim / 256, n_dim);
    gen_ct_f16<<<gct, 256, 0, stream>>>(Ct, n_dim,
        3.14159265358979323846f / (2.0f * (float)n_dim));

    dim3 grid(n_dim / 128, b_dim / 128);
    idct_wmma_gemm<<<grid, 256, 0, stream>>>(Xh, Ct, Y, n_dim);
}
